// DeepseekV3DecoderLayer_70557722738966
// MI455X (gfx1250) — compile-verified, matrix operand order checked by reading, not run on hardware
//
#include <hip/hip_runtime.h>
#include <hip/hip_bf16.h>
#include <math.h>

typedef _Float16 half_t;
typedef __attribute__((ext_vector_type(16))) _Float16 v16h;
typedef __attribute__((ext_vector_type(8)))  _Float16 v8h;
typedef __attribute__((ext_vector_type(8)))  float    v8f;
typedef __attribute__((ext_vector_type(4)))  int      v4i;

#define B_     2
#define S_     2048
#define H_     2048
#define NH_    16
#define QLORA  1536
#define KVLORA 512
#define DNOPE  128
#define DROPE  64
#define DV     128
#define DQK    192
#define INTER_ 8192
#define EPS_   1e-6f
#define M_ROWS (B_*S_)   // 4096

// ---------------------------------------------------------------------------
// CDNA5 async global->LDS copy (ASYNCcnt-tracked), with sync fallback.
// Builtin signature (from clang diagnostic): (as1 v4i*, as3 v4i*, Ii, Ii).
// ---------------------------------------------------------------------------
#ifndef __has_builtin
#define __has_builtin(x) 0
#endif

#if __has_builtin(__builtin_amdgcn_global_load_async_to_lds_b128)
#define USE_ASYNC_LDS 1
#else
#define USE_ASYNC_LDS 0
#endif

__device__ __forceinline__ void async_copy_b128(const half_t* g, half_t* l)
{
#if USE_ASYNC_LDS
    __builtin_amdgcn_global_load_async_to_lds_b128(
        (__attribute__((address_space(1))) v4i*)(v4i*)(g),
        (__attribute__((address_space(3))) v4i*)(v4i*)(l),
        0, 0);
#else
    *(uint4*)l = *(const uint4*)g;
#endif
}

#if USE_ASYNC_LDS
#if __has_builtin(__builtin_amdgcn_s_wait_asynccnt)
#define WAIT_ASYNC() __builtin_amdgcn_s_wait_asynccnt(0)
#else
#define WAIT_ASYNC() asm volatile("s_wait_asynccnt 0x0" ::: "memory")
#endif
#else
#define WAIT_ASYNC() ((void)0)
#endif

// ---------------------------------------------------------------------------
// WMMA fragment load from an LDS row-major tile.
// A-layout per CDNA5 ISA 7.12.2 (16-bit A 16x32): lane<16 holds K{0..7,16..23},
// lane>=16 holds K{8..15,24..31}, row = base + (lane&15). B fragments for
// D = A x B are loaded identically from B^T (N x K) rows.
// ---------------------------------------------------------------------------
__device__ __forceinline__ v16h load_frag(const half_t* __restrict__ base,
                                          int row, int stride, int kbase, int lane) {
    int r   = row + (lane & 15);
    int klo = kbase + ((lane & 16) ? 8 : 0);
    const half_t* p = base + r * stride + klo;
    v8h lo = *(const v8h*)(p);
    v8h hi = *(const v8h*)(p + 16);
    v16h out;
#pragma unroll
    for (int i = 0; i < 8; ++i) { out[i] = lo[i]; out[i + 8] = hi[i]; }
    return out;
}

// ---------------------------------------------------------------------------
// Weight convert: f32 [N,K] -> f16 [Npad,K], zero rows beyond N.
// ---------------------------------------------------------------------------
__global__ void convert_w_kernel(const float* __restrict__ w, half_t* __restrict__ o,
                                 int N, int Npad, int K)
{
    const size_t total = (size_t)Npad * K;
    for (size_t i = (size_t)blockIdx.x * blockDim.x + threadIdx.x; i < total;
         i += (size_t)gridDim.x * blockDim.x) {
        int n = (int)(i / (size_t)K);
        o[i] = (n < N) ? (half_t)w[i] : (half_t)0.f;
    }
}

// ---------------------------------------------------------------------------
// GEMM: C[M,N] = A[M,K](f16) * Wh[Npad,K](f16, transposed-weight) (+ res f32)
// 128x128 tile, K-step 32, 8 waves, each wave 32x64. LDS double-buffered with
// async global->LDS copies overlapping WMMA on the live buffer.
// ---------------------------------------------------------------------------
template<bool OUT_HALF, bool ADD_RES>
__global__ __launch_bounds__(256)
void gemm_f16(const half_t* __restrict__ A, const half_t* __restrict__ Wh,
              void* __restrict__ Cv, const float* __restrict__ R,
              int M, int N, int K)
{
    __shared__ half_t As[2][128 * 40];
    __shared__ half_t Bs[2][128 * 40];

    const int tid   = threadIdx.x;
    const int lane  = tid & 31;
    const int wave  = tid >> 5;
    const int wm    = wave & 3;    // M sub-block *32
    const int wn    = wave >> 2;   // N sub-block *64
    const int tileM = blockIdx.y * 128;
    const int tileN = blockIdx.x * 128;

    v8f acc[2][4];
#pragma unroll
    for (int i = 0; i < 2; ++i)
#pragma unroll
        for (int j = 0; j < 4; ++j)
#pragma unroll
            for (int r = 0; r < 8; ++r) acc[i][j][r] = 0.f;

    const int rl = tid >> 1;          // 0..127
    const int kh = (tid & 1) * 16;    // 0 or 16
    const half_t* Ag = A  + (size_t)(tileM + rl) * K + kh;
    const half_t* Bg = Wh + (size_t)(tileN + rl) * K + kh;
    const int lo = rl * 40 + kh;

    // prologue: stage tile 0 into buffer 0
    async_copy_b128(Ag,     &As[0][lo]);
    async_copy_b128(Ag + 8, &As[0][lo + 8]);
    async_copy_b128(Bg,     &Bs[0][lo]);
    async_copy_b128(Bg + 8, &Bs[0][lo + 8]);
    WAIT_ASYNC();
    __syncthreads();

    const int nk = K >> 5;
    for (int t = 0; t < nk; ++t) {
        const int cur = t & 1;
        if (t + 1 < nk) {   // prefetch next tile into the other buffer
            const half_t* ag = Ag + (size_t)(t + 1) * 32;
            const half_t* bg = Bg + (size_t)(t + 1) * 32;
            async_copy_b128(ag,     &As[cur ^ 1][lo]);
            async_copy_b128(ag + 8, &As[cur ^ 1][lo + 8]);
            async_copy_b128(bg,     &Bs[cur ^ 1][lo]);
            async_copy_b128(bg + 8, &Bs[cur ^ 1][lo + 8]);
        }

        v16h af[2], bf[4];
#pragma unroll
        for (int i = 0; i < 2; ++i) af[i] = load_frag(&As[cur][0], wm * 32 + i * 16, 40, 0, lane);
#pragma unroll
        for (int j = 0; j < 4; ++j) bf[j] = load_frag(&Bs[cur][0], wn * 64 + j * 16, 40, 0, lane);
#pragma unroll
        for (int i = 0; i < 2; ++i)
#pragma unroll
            for (int j = 0; j < 4; ++j)
                acc[i][j] = __builtin_amdgcn_wmma_f32_16x16x32_f16(
                    false, af[i], false, bf[j], (short)0, acc[i][j], false, false);

        WAIT_ASYNC();
        __syncthreads();
    }

    // store: C/D layout -> element r of VGPR group is row r (+8 for lanes>=16)
#pragma unroll
    for (int i = 0; i < 2; ++i)
#pragma unroll
        for (int j = 0; j < 4; ++j)
#pragma unroll
            for (int r = 0; r < 8; ++r) {
                int row = tileM + wm * 32 + i * 16 + r + ((lane & 16) ? 8 : 0);
                int col = tileN + wn * 64 + j * 16 + (lane & 15);
                if (col < N) {
                    float v = acc[i][j][r];
                    if constexpr (ADD_RES) v += R[(size_t)row * N + col];
                    if constexpr (OUT_HALF)
                        ((half_t*)Cv)[(size_t)row * N + col] = (half_t)v;
                    else
                        ((float*)Cv)[(size_t)row * N + col] = v;
                }
            }
}

// ---------------------------------------------------------------------------
// RMSNorm: one block per row; input f32 (row stride may exceed D), output f16.
// ---------------------------------------------------------------------------
__global__ __launch_bounds__(256)
void rmsnorm_kernel(const float* __restrict__ x, int stride, int D,
                    const float* __restrict__ w, half_t* __restrict__ out)
{
    __shared__ float red[256];
    const int row = blockIdx.x;
    const float* xr = x + (size_t)row * stride;
    float s = 0.f;
    for (int i = threadIdx.x; i < D; i += 256) { float v = xr[i]; s += v * v; }
    red[threadIdx.x] = s;
    __syncthreads();
    for (int off = 128; off > 0; off >>= 1) {
        if (threadIdx.x < off) red[threadIdx.x] += red[threadIdx.x + off];
        __syncthreads();
    }
    float rinv = rsqrtf(red[0] / (float)D + EPS_);
    for (int i = threadIdx.x; i < D; i += 256)
        out[(size_t)row * D + i] = (half_t)(w[i] * xr[i] * rinv);
}

// ---------------------------------------------------------------------------
// RoPE helper (interleave-to-half formulation, matching the reference).
// ---------------------------------------------------------------------------
__device__ __forceinline__ float rope_elem(const float* __restrict__ qr, int j, float pos)
{
    int i = j & 31;
    float inv = __expf(-(float)(2 * i) * (1.0f / 64.0f) * 9.210340371976184f); // ln(1e4)
    float f = pos * inv;
    float sn, cs;
    __sincosf(f, &sn, &cs);
    if (j < 32) return qr[2 * j] * cs - qr[2 * j + 1] * sn;
    return qr[2 * i + 1] * cs + qr[2 * i] * sn;
}

// Q pack: [B,S,NH*192] f32 -> [B,NH,S,192] f16 with RoPE on last 64
__global__ void pack_q_kernel(const float* __restrict__ qb, const int* __restrict__ pos,
                              half_t* __restrict__ Qp)
{
    const size_t total = (size_t)B_ * NH_ * S_ * DQK;
    for (size_t idx = (size_t)blockIdx.x * blockDim.x + threadIdx.x; idx < total;
         idx += (size_t)gridDim.x * blockDim.x) {
        int d = (int)(idx % DQK);
        size_t rest = idx / DQK;
        int s = (int)(rest % S_); rest /= S_;
        int h = (int)(rest % NH_);
        int b = (int)(rest / NH_);
        const float* qrow = qb + ((size_t)(b * S_ + s)) * (NH_ * DQK) + h * DQK;
        float v;
        if (d < DNOPE) v = qrow[d];
        else           v = rope_elem(qrow + DNOPE, d - DNOPE, (float)pos[b * S_ + s]);
        Qp[idx] = (half_t)v;
    }
}

// K pack: k_nope from kv_b out, roped k_pe (broadcast over heads) from ckv tail
__global__ void pack_k_kernel(const float* __restrict__ kvb, const float* __restrict__ ckv,
                              const int* __restrict__ pos, half_t* __restrict__ Kp)
{
    const size_t total = (size_t)B_ * NH_ * S_ * DQK;
    for (size_t idx = (size_t)blockIdx.x * blockDim.x + threadIdx.x; idx < total;
         idx += (size_t)gridDim.x * blockDim.x) {
        int d = (int)(idx % DQK);
        size_t rest = idx / DQK;
        int s = (int)(rest % S_); rest /= S_;
        int h = (int)(rest % NH_);
        int b = (int)(rest / NH_);
        float v;
        if (d < DNOPE) {
            v = kvb[((size_t)(b * S_ + s)) * (NH_ * (DNOPE + DV)) + h * (DNOPE + DV) + d];
        } else {
            const float* kr = ckv + (size_t)(b * S_ + s) * (KVLORA + DROPE) + KVLORA;
            v = rope_elem(kr, d - DNOPE, (float)pos[b * S_ + s]);
        }
        Kp[idx] = (half_t)v;
    }
}

// V pack, TRANSPOSED: [B,S,NH*256] f32 -> [B,NH,DV,S] f16
__global__ void pack_v_kernel(const float* __restrict__ kvb, half_t* __restrict__ Vp)
{
    const size_t total = (size_t)B_ * NH_ * DV * S_;
    for (size_t idx = (size_t)blockIdx.x * blockDim.x + threadIdx.x; idx < total;
         idx += (size_t)gridDim.x * blockDim.x) {
        int s = (int)(idx % S_);
        size_t rest = idx / S_;
        int n = (int)(rest % DV); rest /= DV;
        int h = (int)(rest % NH_);
        int b = (int)(rest / NH_);
        float v = kvb[((size_t)(b * S_ + s)) * (NH_ * (DNOPE + DV)) + h * (DNOPE + DV) + DNOPE + n];
        Vp[idx] = (half_t)v;
    }
}

// ---------------------------------------------------------------------------
// Flash attention (causal): 64 q-rows per block, 64-wide k tiles, D_QK=192,
// D_V=128. 8 waves: wave w owns O rows [(w>>1)*16,+16), cols [(w&1)*64,+64).
// K/V tiles staged with async global->LDS copies.
// ---------------------------------------------------------------------------
__global__ __launch_bounds__(256)
void flash_kernel(const half_t* __restrict__ Qp, const half_t* __restrict__ Kp,
                  const half_t* __restrict__ Vp, half_t* __restrict__ Out)
{
    extern __shared__ char smem[];
    half_t* Qs   = (half_t*)smem;            // 64 x 200
    half_t* Ks   = Qs + 64 * 200;            // 64 x 200
    half_t* Vt   = Ks + 64 * 200;            // 128 x 72  (V transposed: [n][k])
    half_t* Ps   = Vt + 128 * 72;            // 64 x 72
    float*  Sl   = (float*)(Ps + 64 * 72);   // 64 x 68
    float*  mrow = Sl + 64 * 68;
    float*  lrow = mrow + 64;
    float*  arow = lrow + 64;

    const int tid  = threadIdx.x;
    const int lane = tid & 31;
    const int wave = tid >> 5;
    const int qt = blockIdx.x;
    const int h  = blockIdx.y;
    const int b  = blockIdx.z;
    const int q0 = qt * 64;
    const size_t bh = (size_t)b * NH_ + h;
    const float scale = rsqrtf((float)DQK);

    // load Q tile once (async b128 chunks)
    const half_t* qg = Qp + (bh * S_ + q0) * DQK;
    for (int c = tid; c < 64 * 24; c += 256) {
        int r = c / 24, off = (c % 24) * 8;
        async_copy_b128(qg + (size_t)r * DQK + off, Qs + r * 200 + off);
    }
    if (tid < 64) { mrow[tid] = -3.0e38f; lrow[tid] = 0.f; }

    const int obase = (wave >> 1) * 16;
    const int jblk  = wave & 1;
    v8f accO[4];
#pragma unroll
    for (int j = 0; j < 4; ++j)
#pragma unroll
        for (int r = 0; r < 8; ++r) accO[j][r] = 0.f;

    WAIT_ASYNC();
    __syncthreads();

    for (int kt = 0; kt <= qt; ++kt) {
        const int k0 = kt * 64;
        const half_t* kg = Kp + (bh * S_ + k0) * DQK;
        for (int c = tid; c < 64 * 24; c += 256) {
            int r = c / 24, off = (c % 24) * 8;
            async_copy_b128(kg + (size_t)r * DQK + off, Ks + r * 200 + off);
        }
        const half_t* vg = Vp + bh * DV * S_ + k0;   // [bh][n][s]
        for (int c = tid; c < 128 * 8; c += 256) {
            int n = c / 8, off = (c % 8) * 8;
            async_copy_b128(vg + (size_t)n * S_ + off, Vt + n * 72 + off);
        }
        WAIT_ASYNC();
        __syncthreads();

        // S tile: wave computes rows obase..+16, cols (jblk*32)..+32
        {
            const int scol = jblk * 32;
            v8f sacc[2];
#pragma unroll
            for (int j = 0; j < 2; ++j)
#pragma unroll
                for (int r = 0; r < 8; ++r) sacc[j][r] = 0.f;
#pragma unroll
            for (int ks = 0; ks < DQK; ks += 32) {
                v16h a = load_frag(Qs, obase, 200, ks, lane);
#pragma unroll
                for (int j = 0; j < 2; ++j) {
                    v16h bb = load_frag(Ks, scol + j * 16, 200, ks, lane);
                    sacc[j] = __builtin_amdgcn_wmma_f32_16x16x32_f16(
                        false, a, false, bb, (short)0, sacc[j], false, false);
                }
            }
#pragma unroll
            for (int j = 0; j < 2; ++j)
#pragma unroll
                for (int r = 0; r < 8; ++r) {
                    int rr = obase + r + ((lane & 16) ? 8 : 0);
                    int cc = scol + j * 16 + (lane & 15);
                    float v = sacc[j][r] * scale;
                    if (k0 + cc > q0 + rr) v = -1.0e9f;  // causal NEG_INF
                    Sl[rr * 68 + cc] = v;
                }
        }
        __syncthreads();

        // online softmax update (one thread per row)
        if (tid < 64) {
            int r = tid;
            float mold = mrow[r];
            float mx = mold;
            for (int c2 = 0; c2 < 64; ++c2) mx = fmaxf(mx, Sl[r * 68 + c2]);
            float lsum = 0.f;
            for (int c2 = 0; c2 < 64; ++c2) {
                float p = __expf(Sl[r * 68 + c2] - mx);
                Ps[r * 72 + c2] = (half_t)p;
                lsum += p;
            }
            float al = __expf(mold - mx);
            lrow[r] = lrow[r] * al + lsum;
            mrow[r] = mx;
            arow[r] = al;
        }
        __syncthreads();

        // rescale running O and accumulate P * V
#pragma unroll
        for (int j = 0; j < 4; ++j)
#pragma unroll
            for (int r = 0; r < 8; ++r) {
                int rr = obase + r + ((lane & 16) ? 8 : 0);
                accO[j][r] *= arow[rr];
            }
#pragma unroll
        for (int ks = 0; ks < 64; ks += 32) {
            v16h pa = load_frag(Ps, obase, 72, ks, lane);
#pragma unroll
            for (int j = 0; j < 4; ++j) {
                v16h vb = load_frag(Vt, jblk * 64 + j * 16, 72, ks, lane);
                accO[j] = __builtin_amdgcn_wmma_f32_16x16x32_f16(
                    false, pa, false, vb, (short)0, accO[j], false, false);
            }
        }
        __syncthreads();
    }

    // normalize and store to [B, S, NH*DV] f16
#pragma unroll
    for (int j = 0; j < 4; ++j)
#pragma unroll
        for (int r = 0; r < 8; ++r) {
            int rr = obase + r + ((lane & 16) ? 8 : 0);
            int cc = jblk * 64 + j * 16 + (lane & 15);
            float v = accO[j][r] / lrow[rr];
            Out[((size_t)(b * S_ + q0 + rr)) * (NH_ * DV) + h * DV + cc] = (half_t)v;
        }
}

// silu(g) * u, in place into g
__global__ void silu_mul_kernel(half_t* __restrict__ g, const half_t* __restrict__ u, size_t n)
{
    for (size_t i = (size_t)blockIdx.x * blockDim.x + threadIdx.x; i < n;
         i += (size_t)gridDim.x * blockDim.x) {
        float gv = (float)g[i];
        float uv = (float)u[i];
        float sv = gv / (1.f + __expf(-gv));
        g[i] = (half_t)(sv * uv);
    }
}

// ---------------------------------------------------------------------------
extern "C" void kernel_launch(void* const* d_in, const int* in_sizes, int n_in,
                              void* d_out, int out_size, void* d_ws, size_t ws_size,
                              hipStream_t stream)
{
    const float* hidden    = (const float*)d_in[0];
    const int*   pos       = (const int*)  d_in[1];
    const float* in_ln_w   = (const float*)d_in[2];
    const float* post_ln_w = (const float*)d_in[3];
    const float* q_a_w     = (const float*)d_in[4];
    const float* q_a_ln_w  = (const float*)d_in[5];
    const float* q_b_w     = (const float*)d_in[6];
    const float* kv_a_w    = (const float*)d_in[7];
    const float* kv_a_ln_w = (const float*)d_in[8];
    const float* kv_b_w    = (const float*)d_in[9];
    const float* o_w       = (const float*)d_in[10];
    const float* gate_w    = (const float*)d_in[11];
    const float* up_w      = (const float*)d_in[12];
    const float* down_w    = (const float*)d_in[13];
    float* out = (float*)d_out;

    // workspace layout with lifetime-based reuse; peak = 288 MiB
    char* ws = (char*)d_ws;
    half_t* x16    = (half_t*)(ws + 0);             // dead after kv_a gemm
    float*  qa     = (float*) (ws + 16777216ULL);   // dead after qa16
    half_t* qa16   = (half_t*)(ws + 41943040ULL);   // dead after q_b gemm
    float*  qb     = (float*) (ws + 54525952ULL);   // dead after pack_q
    float*  ckv    = (float*) (ws + 104857600ULL);  // dead after pack_k
    half_t* ckv16  = (half_t*)(ws + 114294784ULL);  // dead after kv_b gemm
    float*  kvb    = (float*) (ws + 118489088ULL);  // dead after packs
    half_t* Qp     = (half_t*)(ws + 0);             // reuses x16/qa (dead)
    half_t* Kp     = (half_t*)(ws + 25165824ULL);   // reuses qa/qa16 (dead)
    half_t* Vp     = (half_t*)(ws + 50331648ULL);   // reuses qb head (after pack_q)
    half_t* attn16 = (half_t*)(ws + 67108864ULL);
    float*  x1     = (float*) (ws + 83886080ULL);
    half_t* x1_16  = (half_t*)(ws + 117440512ULL);
    half_t* g16    = (half_t*)(ws + 134217728ULL);
    half_t* u16    = (half_t*)(ws + 201326592ULL);
    half_t* wc     = (half_t*)(ws + 268435456ULL);  // converted weights, 32 MiB slot

    const int M = M_ROWS;
    dim3 blk(256);

    // 1) x = rms_norm(hidden, in_ln_w)  -> f16
    rmsnorm_kernel<<<M, blk, 0, stream>>>(hidden, H_, H_, in_ln_w, x16);

    // 2) qa = x @ q_a_w^T
    convert_w_kernel<<<2048, blk, 0, stream>>>(q_a_w, wc, QLORA, QLORA, H_);
    gemm_f16<false, false><<<dim3(QLORA / 128, M / 128), blk, 0, stream>>>(
        x16, wc, qa, nullptr, M, QLORA, H_);

    // 3) qa16 = rms_norm(qa, q_a_ln_w)
    rmsnorm_kernel<<<M, blk, 0, stream>>>(qa, QLORA, QLORA, q_a_ln_w, qa16);

    // 4) qb = qa16 @ q_b_w^T
    convert_w_kernel<<<2048, blk, 0, stream>>>(q_b_w, wc, NH_ * DQK, NH_ * DQK, QLORA);
    gemm_f16<false, false><<<dim3((NH_ * DQK) / 128, M / 128), blk, 0, stream>>>(
        qa16, wc, qb, nullptr, M, NH_ * DQK, QLORA);

    // 5) ckv = x @ kv_a_w^T  (N=576 -> pad to 640)
    convert_w_kernel<<<2048, blk, 0, stream>>>(kv_a_w, wc, KVLORA + DROPE, 640, H_);
    gemm_f16<false, false><<<dim3(640 / 128, M / 128), blk, 0, stream>>>(
        x16, wc, ckv, nullptr, M, KVLORA + DROPE, H_);

    // 6) ckv16 = rms_norm(ckv[:, :512], kv_a_ln_w)
    rmsnorm_kernel<<<M, blk, 0, stream>>>(ckv, KVLORA + DROPE, KVLORA, kv_a_ln_w, ckv16);

    // 7) kvb = ckv16 @ kv_b_w^T
    convert_w_kernel<<<2048, blk, 0, stream>>>(kv_b_w, wc, NH_ * (DNOPE + DV), NH_ * (DNOPE + DV), KVLORA);
    gemm_f16<false, false><<<dim3((NH_ * (DNOPE + DV)) / 128, M / 128), blk, 0, stream>>>(
        ckv16, wc, kvb, nullptr, M, NH_ * (DNOPE + DV), KVLORA);

    // 8) pack Q (rope), K (nope|rope), V (transposed)
    pack_q_kernel<<<4096, blk, 0, stream>>>(qb, pos, Qp);
    pack_k_kernel<<<4096, blk, 0, stream>>>(kvb, ckv, pos, Kp);
    pack_v_kernel<<<4096, blk, 0, stream>>>(kvb, Vp);

    // 9) flash attention
    {
        size_t smem = (size_t)(64 * 200 + 64 * 200 + 128 * 72 + 64 * 72) * 2 +
                      (size_t)(64 * 68 + 3 * 64) * 4;   // 97024 bytes
        flash_kernel<<<dim3(S_ / 64, NH_, B_), blk, smem, stream>>>(Qp, Kp, Vp, attn16);
    }

    // 10) x1 = attn @ o_w^T + hidden  (residual)
    convert_w_kernel<<<2048, blk, 0, stream>>>(o_w, wc, H_, H_, NH_ * DV);
    gemm_f16<false, true><<<dim3(H_ / 128, M / 128), blk, 0, stream>>>(
        attn16, wc, x1, hidden, M, H_, NH_ * DV);

    // 11) x1_16 = rms_norm(x1, post_ln_w)
    rmsnorm_kernel<<<M, blk, 0, stream>>>(x1, H_, H_, post_ln_w, x1_16);

    // 12) gate & up projections (f16 out)
    convert_w_kernel<<<4096, blk, 0, stream>>>(gate_w, wc, INTER_, INTER_, H_);
    gemm_f16<true, false><<<dim3(INTER_ / 128, M / 128), blk, 0, stream>>>(
        x1_16, wc, g16, nullptr, M, INTER_, H_);
    convert_w_kernel<<<4096, blk, 0, stream>>>(up_w, wc, INTER_, INTER_, H_);
    gemm_f16<true, false><<<dim3(INTER_ / 128, M / 128), blk, 0, stream>>>(
        x1_16, wc, u16, nullptr, M, INTER_, H_);

    // 13) h = silu(g) * u  (in place into g16)
    silu_mul_kernel<<<4096, blk, 0, stream>>>(g16, u16, (size_t)M * INTER_);

    // 14) out = h @ down_w^T + x1  (residual, f32 -> d_out)
    convert_w_kernel<<<4096, blk, 0, stream>>>(down_w, wc, H_, H_, INTER_);
    gemm_f16<false, true><<<dim3(H_ / 128, M / 128), blk, 0, stream>>>(
        g16, wc, out, x1, M, H_, INTER_);
}
